// GraphAttentionNetwork_49555332661699
// MI455X (gfx1250) — matrix operand order repaired and checked by reading, not compiled, not run on hardware
//
#include <hip/hip_runtime.h>

#define IN_C     128
#define HIDC     256
#define NHEADS   4
#define HDIM     64
#define NLAYERS  4
#define NGRAPHS  64
#define OUTC     32

typedef __bf16 bf16_t;
typedef __attribute__((ext_vector_type(16))) __bf16 v16bf;
typedef __attribute__((ext_vector_type(8)))  __bf16 v8bf;
typedef __attribute__((ext_vector_type(8)))  float  v8f;

enum { EP_RELU = 1, EP_ACCUM = 2 };

// ---------------------------------------------------------------------------
// bf16 WMMA GEMM with LDS-staged B panel + double buffering.
//   C[M x Nc] = A[M x K] @ W[K x Nc] (+bias)(+C)(relu)(+resid)
// Block = 8 waves sharing one N-panel (NW = 16*NSUB wide). Each wave computes
// a 32 x NW tile (2 m-subtiles x NSUB n-subtiles = 2*NSUB WMMA accumulators).
// B chunk (32 x NW) double-buffered in LDS; padded row stride (16B) makes the
// per-element column fragment reads bank-conflict-free.
// Out-of-range m-subtiles are CLAMPED (not branched): the K-loop is fully
// branch-free so WMMA always runs with EXEC all-1s and the scheduler can
// pipeline loads against WMMAs; results of clamped tiles are dropped in the
// epilogue.
// ---------------------------------------------------------------------------
template<int NSUB>
__global__ __launch_bounds__(256)
void wmma_gemm_lds_kernel(const bf16_t* __restrict__ A, const bf16_t* __restrict__ W,
                          const float* __restrict__ bias, const float* __restrict__ resid,
                          float* __restrict__ C, bf16_t* __restrict__ Cb,
                          int M, int K, int Nc, int flags)
{
    constexpr int NW     = 16 * NSUB;   // N-panel width
    constexpr int LSTR   = NW + 8;      // LDS row stride (elements); +16B pad
    constexpr int CHUNKS = 4 * NW;      // 8-elem chunks per 32xNW tile
    __shared__ bf16_t smem[2][32 * LSTR];

    const int tid  = threadIdx.x;
    const int lane = tid & 31;
    const int wv   = tid >> 5;                       // wave in block (0..7)
    const int nUnits = Nc / NW;
    const int mb = blockIdx.x / nUnits;
    const int nu = blockIdx.x - mb * nUnits;
    const int n0 = nu * NW;
    const int m0 = mb * 256 + wv * 32;               // two 16-row subtiles

    const int mrow = lane & 15;
    const int half = lane >> 4;

    const bool val0 = (m0)      < M;                 // M is a multiple of 16
    const bool val1 = (m0 + 16) < M;
    const int  r0   = val0 ? m0      : 0;            // clamped row bases
    const int  r1   = val1 ? m0 + 16 : 0;

    v8f acc[2][NSUB] = {};

    // staging role: one 8-elem (16B) chunk per thread (all threads for NSUB=4)
    const int srow = tid / (NW / 8);
    const int scol = (tid - srow * (NW / 8)) * 8;

    const bf16_t* aRow0 = A + (size_t)(r0 + mrow) * K + half * 8;
    const bf16_t* aRow1 = A + (size_t)(r1 + mrow) * K + half * 8;

    // prologue: stage K-step 0; prefetch A fragments for K-step 0
    if (CHUNKS >= 256 || tid < CHUNKS) {
        v8bf g = *(const v8bf*)(W + (size_t)srow * Nc + n0 + scol);
        *(v8bf*)&smem[0][srow * LSTR + scol] = g;
    }
    v16bf a0, a1;
    *((v8bf*)&a0)       = *(const v8bf*)(aRow0);
    *(((v8bf*)&a0) + 1) = *(const v8bf*)(aRow0 + 16);
    *((v8bf*)&a1)       = *(const v8bf*)(aRow1);
    *(((v8bf*)&a1) + 1) = *(const v8bf*)(aRow1 + 16);

    const int S = K >> 5;
    for (int s = 0; s < S; ++s) {
        __syncthreads();                             // stage(s) visible; prior reads done
        v16bf a0n, a1n;
        if (s + 1 < S) {                             // scalar (wave-uniform) branch
            if (CHUNKS >= 256 || tid < CHUNKS) {
                v8bf g = *(const v8bf*)(W + (size_t)((s + 1) * 32 + srow) * Nc + n0 + scol);
                *(v8bf*)&smem[(s + 1) & 1][srow * LSTR + scol] = g;
            }
            const int kn = (s + 1) * 32;
            *((v8bf*)&a0n)       = *(const v8bf*)(aRow0 + kn);
            *(((v8bf*)&a0n) + 1) = *(const v8bf*)(aRow0 + kn + 16);
            *((v8bf*)&a1n)       = *(const v8bf*)(aRow1 + kn);
            *(((v8bf*)&a1n) + 1) = *(const v8bf*)(aRow1 + kn + 16);
        }

        const bf16_t* bbase = &smem[s & 1][(size_t)(half * 8) * LSTR + (lane & 15)];
#pragma unroll
        for (int t = 0; t < NSUB; ++t) {
            v16bf b;
#pragma unroll
            for (int i = 0; i < 8; ++i) {
                b[i]     = bbase[i * LSTR + t * 16];
                b[8 + i] = bbase[(16 + i) * LSTR + t * 16];
            }
            acc[0][t] = __builtin_amdgcn_wmma_f32_16x16x32_bf16(
                            false, a0, false, b, (short)0, acc[0][t], false, false);
            acc[1][t] = __builtin_amdgcn_wmma_f32_16x16x32_bf16(
                            false, a1, false, b, (short)0, acc[1][t], false, false);
        }
        if (s + 1 < S) { a0 = a0n; a1 = a1n; }
    }

    // epilogue (clamped tiles dropped here)
    const int c0 = n0 + (lane & 15);
#pragma unroll
    for (int u = 0; u < 2; ++u) {
        if (u == 0 ? !val0 : !val1) continue;
        const int rBase = m0 + u * 16 + half * 8;
#pragma unroll
        for (int t = 0; t < NSUB; ++t) {
            const int col = c0 + t * 16;
            const float bv = bias ? bias[col] : 0.0f;
#pragma unroll
            for (int i = 0; i < 8; ++i) {
                const size_t idx = (size_t)(rBase + i) * Nc + col;
                float v = acc[u][t][i] + bv;
                if (flags & EP_ACCUM) v += C[idx];
                if (flags & EP_RELU)  v = v > 0.0f ? v : 0.0f;
                if (resid) v += resid[idx];
                C[idx] = v;
                if (Cb) Cb[idx] = (bf16_t)v;
            }
        }
    }
}

// ---------------------------------------------------------------------------
// LayerNorm over HIDC=256, one wave per row, bf16 output.
// ---------------------------------------------------------------------------
__global__ __launch_bounds__(256)
void ln_bf16_kernel(const float* __restrict__ X, const float* __restrict__ g,
                    const float* __restrict__ beta, bf16_t* __restrict__ Y, int M)
{
    const int lane = threadIdx.x & 31;
    const int row  = (int)((blockIdx.x * blockDim.x + threadIdx.x) >> 5);
    if (row >= M) return;
    const float* x = X + (size_t)row * HIDC;
    float v[8];
    float s = 0.0f;
#pragma unroll
    for (int j = 0; j < 8; ++j) { v[j] = x[lane + 32 * j]; s += v[j]; }
#pragma unroll
    for (int o = 16; o > 0; o >>= 1) s += __shfl_xor(s, o, 32);
    const float mu = s * (1.0f / HIDC);
    float vs = 0.0f;
#pragma unroll
    for (int j = 0; j < 8; ++j) { float d = v[j] - mu; vs += d * d; }
#pragma unroll
    for (int o = 16; o > 0; o >>= 1) vs += __shfl_xor(vs, o, 32);
    const float inv = rsqrtf(vs * (1.0f / HIDC) + 1e-5f);
    bf16_t* y = Y + (size_t)row * HIDC;
#pragma unroll
    for (int j = 0; j < 8; ++j) {
        const int c = lane + 32 * j;
        y[c] = (bf16_t)((v[j] - mu) * inv * g[c] + beta[c]);
    }
}

// ---------------------------------------------------------------------------
// Elementwise helpers
// ---------------------------------------------------------------------------
__global__ void f32_to_bf16_kernel(const float* __restrict__ s, bf16_t* __restrict__ d, size_t n)
{
    size_t i = (size_t)blockIdx.x * blockDim.x + threadIdx.x;
    if (i < n) d[i] = (bf16_t)s[i];
}

__global__ void zero_kernel(float* __restrict__ p, size_t n)
{
    size_t i = (size_t)blockIdx.x * blockDim.x + threadIdx.x;
    if (i < n) p[i] = 0.0f;
}

__global__ void pos_add_kernel(float* __restrict__ h, bf16_t* __restrict__ hb,
                               const float* __restrict__ pos_emb,
                               const int* __restrict__ pos_idx, int n)
{
    size_t i = (size_t)blockIdx.x * blockDim.x + threadIdx.x;
    if (i >= (size_t)n * HIDC) return;
    const int node = (int)(i >> 8);
    const int c    = (int)(i & 255);
    float v = h[i] + pos_emb[(size_t)pos_idx[node] * HIDC + c];
    h[i]  = v;
    hb[i] = (bf16_t)v;
}

// ---------------------------------------------------------------------------
// Ordered-uint encoding for float atomic max
// ---------------------------------------------------------------------------
__device__ __forceinline__ unsigned f2ord(float f)
{
    unsigned u = __float_as_uint(f);
    return (u & 0x80000000u) ? ~u : (u | 0x80000000u);
}
__device__ __forceinline__ float ord2f(unsigned u)
{
    return __uint_as_float((u & 0x80000000u) ? (u ^ 0x80000000u) : ~u);
}

// ---------------------------------------------------------------------------
// Edge pass 1: scores[e][h] = (Q[row] . K[col])_h / 8 ; atomicMax per (row,h)
// One wave per edge; 8 lanes per head.
// ---------------------------------------------------------------------------
__global__ __launch_bounds__(256)
void edge_score_kernel(const float* __restrict__ Q, const float* __restrict__ Kf,
                       const int* __restrict__ erow, const int* __restrict__ ecol,
                       float* __restrict__ scores, unsigned* __restrict__ mmax, int E)
{
    const int lane = threadIdx.x & 31;
    const int e    = (int)((blockIdx.x * blockDim.x + threadIdx.x) >> 5);
    if (e >= E) return;
    const int r = erow[e], c = ecol[e];
    const int head = lane >> 3, t = lane & 7;
    const float* q = Q  + (size_t)r * HIDC + head * HDIM + t;
    const float* k = Kf + (size_t)c * HIDC + head * HDIM + t;
    float s = 0.0f;
#pragma unroll
    for (int j = 0; j < 8; ++j) s += q[8 * j] * k[8 * j];
    s += __shfl_xor(s, 1, 32);
    s += __shfl_xor(s, 2, 32);
    s += __shfl_xor(s, 4, 32);
    if (t == 0) {
        s *= 0.125f;                      // 1/sqrt(HDIM)
        scores[(size_t)e * NHEADS + head] = s;
        atomicMax(&mmax[(size_t)r * NHEADS + head], f2ord(s));
    }
}

// ---------------------------------------------------------------------------
// Edge pass 2: ex = exp(s - m); denom += ex; agg[row] += ex * V[col]
// ---------------------------------------------------------------------------
__global__ __launch_bounds__(256)
void edge_msg_kernel(const float* __restrict__ V,
                     const int* __restrict__ erow, const int* __restrict__ ecol,
                     const float* __restrict__ scores, const unsigned* __restrict__ mmax,
                     float* __restrict__ denom, float* __restrict__ agg, int E)
{
    const int lane = threadIdx.x & 31;
    const int e    = (int)((blockIdx.x * blockDim.x + threadIdx.x) >> 5);
    if (e >= E) return;
    const int r = erow[e], c = ecol[e];
    const int head = lane >> 3, t = lane & 7;
    const float s  = scores[(size_t)e * NHEADS + head];
    const float m  = ord2f(mmax[(size_t)r * NHEADS + head]);
    const float ex = __expf(s - m);
    if (t == 0) atomicAdd(&denom[(size_t)r * NHEADS + head], ex);
    const float* v = V   + (size_t)c * HIDC + head * HDIM + t;
    float*       a = agg + (size_t)r * HIDC + head * HDIM + t;
#pragma unroll
    for (int j = 0; j < 8; ++j) atomicAdd(&a[8 * j], ex * v[8 * j]);
}

__global__ void agg_norm_kernel(const float* __restrict__ agg, const float* __restrict__ denom,
                                bf16_t* __restrict__ aggb, int n)
{
    size_t i = (size_t)blockIdx.x * blockDim.x + threadIdx.x;
    if (i >= (size_t)n * HIDC) return;
    const int node = (int)(i >> 8);
    const int head = (int)((i & 255) >> 6);
    const float d  = denom[(size_t)node * NHEADS + head];
    aggb[i] = (bf16_t)(d > 0.0f ? agg[i] / d : 0.0f);
}

// ---------------------------------------------------------------------------
// Pooling
// ---------------------------------------------------------------------------
__global__ void pool_sum_kernel(const float* __restrict__ h, const int* __restrict__ batch,
                                float* __restrict__ pooled, float* __restrict__ cnt, int n)
{
    size_t i = (size_t)blockIdx.x * blockDim.x + threadIdx.x;
    if (i >= (size_t)n * HIDC) return;
    const int node = (int)(i >> 8);
    const int c    = (int)(i & 255);
    const int g    = batch[node];
    atomicAdd(&pooled[(size_t)g * HIDC + c], h[i]);
    if (c == 0) atomicAdd(&cnt[g], 1.0f);
}

__global__ void pool_finish_kernel(float* __restrict__ pooled, const float* __restrict__ cnt,
                                   bf16_t* __restrict__ pooledb)
{
    int i = (int)(blockIdx.x * blockDim.x + threadIdx.x);
    if (i >= NGRAPHS * HIDC) return;
    const float v = pooled[i] / fmaxf(cnt[i >> 8], 1.0f);
    pooled[i]  = v;
    pooledb[i] = (bf16_t)v;
}

// ---------------------------------------------------------------------------
// Host launcher
// ---------------------------------------------------------------------------
static inline int ceil_div(long long a, long long b) { return (int)((a + b - 1) / b); }

extern "C" void kernel_launch(void* const* d_in, const int* in_sizes, int n_in,
                              void* d_out, int out_size, void* d_ws, size_t ws_size,
                              hipStream_t stream)
{
    (void)n_in; (void)out_size; (void)ws_size;

    const int N = in_sizes[0] / IN_C;       // 50000
    const int E = in_sizes[1] / 2;          // 800000

    const float* x        = (const float*)d_in[0];
    const int*   erow     = (const int*)d_in[1];
    const int*   ecol     = erow + E;
    const int*   pos_idx  = (const int*)d_in[2];
    const int*   batch    = (const int*)d_in[3];
    const float* enc_w1   = (const float*)d_in[4];
    const float* enc_b1   = (const float*)d_in[5];
    const float* enc_g    = (const float*)d_in[6];
    const float* enc_beta = (const float*)d_in[7];
    const float* enc_w2   = (const float*)d_in[8];
    const float* enc_b2   = (const float*)d_in[9];
    const float* pos_emb  = (const float*)d_in[10];
    const float* Wq       = (const float*)d_in[11];
    const float* bq       = (const float*)d_in[12];
    const float* Wk       = (const float*)d_in[13];
    const float* bk       = (const float*)d_in[14];
    const float* Wv       = (const float*)d_in[15];
    const float* bv       = (const float*)d_in[16];
    const float* mlp_w1   = (const float*)d_in[17];
    const float* mlp_b1   = (const float*)d_in[18];
    const float* mlp_g    = (const float*)d_in[19];
    const float* mlp_beta = (const float*)d_in[20];
    const float* mlp_w2   = (const float*)d_in[21];
    const float* mlp_b2   = (const float*)d_in[22];
    const float* cls_w1   = (const float*)d_in[23];
    const float* cls_b1   = (const float*)d_in[24];
    const float* cls_g    = (const float*)d_in[25];
    const float* cls_beta = (const float*)d_in[26];
    const float* cls_w2   = (const float*)d_in[27];
    const float* cls_b2   = (const float*)d_in[28];

    // ---- workspace carve-out (256B aligned slabs) ----
    char* wsp = (char*)d_ws;
    auto alloc = [&](size_t bytes) -> void* {
        void* p = (void*)wsp;
        wsp += (bytes + 255) & ~(size_t)255;
        return p;
    };
    const size_t NH = (size_t)N * HIDC;

    float*  h    = (float*)alloc(NH * 4);
    float*  Qf   = (float*)alloc(NH * 4);      // reused as MLP hidden t1
    float*  Kf   = (float*)alloc(NH * 4);
    float*  Vf   = (float*)alloc(NH * 4);
    float*  agg  = (float*)alloc(NH * 4);
    float*  denom = (float*)alloc((size_t)N * NHEADS * 4);
    unsigned* mmax = (unsigned*)alloc((size_t)N * NHEADS * 4);
    float*  scores = (float*)alloc((size_t)E * NHEADS * 4);
    bf16_t* hb   = (bf16_t*)alloc(NH * 2);
    bf16_t* t1b  = (bf16_t*)alloc(NH * 2);
    bf16_t* aggb = (bf16_t*)alloc(NH * 2);
    bf16_t* xb   = (bf16_t*)alloc((size_t)N * IN_C * 2);
    float*  pooled  = (float*)alloc((size_t)NGRAPHS * HIDC * 4);
    float*  cnt     = (float*)alloc((size_t)NGRAPHS * 4);
    bf16_t* pooledb = (bf16_t*)alloc((size_t)NGRAPHS * HIDC * 2);
    float*  c1      = (float*)alloc((size_t)NGRAPHS * HIDC * 4);
    bf16_t* c1b     = (bf16_t*)alloc((size_t)NGRAPHS * HIDC * 2);
    // bf16 weights
    bf16_t* enc_w1b = (bf16_t*)alloc((size_t)IN_C * HIDC * 2);
    bf16_t* enc_w2b = (bf16_t*)alloc((size_t)HIDC * HIDC * 2);
    bf16_t* Wqb  = (bf16_t*)alloc((size_t)NLAYERS * HIDC * HIDC * 2);
    bf16_t* Wkb  = (bf16_t*)alloc((size_t)NLAYERS * HIDC * HIDC * 2);
    bf16_t* Wvb  = (bf16_t*)alloc((size_t)NLAYERS * HIDC * HIDC * 2);
    bf16_t* mw1b = (bf16_t*)alloc((size_t)NLAYERS * 2 * HIDC * HIDC * 2);
    bf16_t* mw2b = (bf16_t*)alloc((size_t)NLAYERS * HIDC * HIDC * 2);
    bf16_t* cw1b = (bf16_t*)alloc((size_t)HIDC * HIDC * 2);
    bf16_t* cw2b = (bf16_t*)alloc((size_t)HIDC * OUTC * 2);

    const int TB = 256;
    auto cvt = [&](const float* s, bf16_t* d, size_t n) {
        f32_to_bf16_kernel<<<ceil_div((long long)n, TB), TB, 0, stream>>>(s, d, n);
    };
    auto zero = [&](float* p, size_t n) {
        zero_kernel<<<ceil_div((long long)n, TB), TB, 0, stream>>>(p, n);
    };
    // gemm: block covers 256 rows x 64 cols (8 waves, 32x64 per wave)
    auto gemm4 = [&](const bf16_t* A, const bf16_t* W, const float* bias, const float* resid,
                     float* C, bf16_t* Cb, int M, int K, int Nc, int flags) {
        const int blocks = ceil_div(M, 256) * (Nc / 64);
        wmma_gemm_lds_kernel<4><<<blocks, TB, 0, stream>>>(A, W, bias, resid, C, Cb, M, K, Nc, flags);
    };

    // ---- weight/input conversion (every call; deterministic) ----
    cvt(x,      xb,      (size_t)N * IN_C);
    cvt(enc_w1, enc_w1b, (size_t)IN_C * HIDC);
    cvt(enc_w2, enc_w2b, (size_t)HIDC * HIDC);
    cvt(Wq,     Wqb,     (size_t)NLAYERS * HIDC * HIDC);
    cvt(Wk,     Wkb,     (size_t)NLAYERS * HIDC * HIDC);
    cvt(Wv,     Wvb,     (size_t)NLAYERS * HIDC * HIDC);
    cvt(mlp_w1, mw1b,    (size_t)NLAYERS * 2 * HIDC * HIDC);
    cvt(mlp_w2, mw2b,    (size_t)NLAYERS * HIDC * HIDC);
    cvt(cls_w1, cw1b,    (size_t)HIDC * HIDC);
    cvt(cls_w2, cw2b,    (size_t)HIDC * OUTC);

    const int edgeBlocks = ceil_div(E, 8);          // 1 wave per edge
    const int lnBlocksN  = ceil_div(N, 8);          // 1 wave per row

    // ---- encoder: h = LN(relu(x@W1+b1)) @ W2 + b2 ; h += pos_emb[pos_idx] ----
    gemm4(xb, enc_w1b, enc_b1, nullptr, Qf, nullptr, N, IN_C, HIDC, EP_RELU);
    ln_bf16_kernel<<<lnBlocksN, TB, 0, stream>>>(Qf, enc_g, enc_beta, t1b, N);
    gemm4(t1b, enc_w2b, enc_b2, nullptr, h, nullptr, N, HIDC, HIDC, 0);
    pos_add_kernel<<<ceil_div((long long)NH, TB), TB, 0, stream>>>(h, hb, pos_emb, pos_idx, N);

    // ---- attention layers ----
    for (int l = 0; l < NLAYERS; ++l) {
        const size_t wOff  = (size_t)l * HIDC * HIDC;
        const size_t w1Off = (size_t)l * 2 * HIDC * HIDC;
        const size_t bOff  = (size_t)l * HIDC;

        gemm4(hb, Wqb + wOff, bq + bOff, nullptr, Qf, nullptr, N, HIDC, HIDC, 0);
        gemm4(hb, Wkb + wOff, bk + bOff, nullptr, Kf, nullptr, N, HIDC, HIDC, 0);
        gemm4(hb, Wvb + wOff, bv + bOff, nullptr, Vf, nullptr, N, HIDC, HIDC, 0);

        zero(agg, NH);
        zero(denom, (size_t)N * NHEADS);
        zero((float*)mmax, (size_t)N * NHEADS);     // 0 bits == ordered -inf side

        edge_score_kernel<<<edgeBlocks, TB, 0, stream>>>(Qf, Kf, erow, ecol, scores, mmax, E);
        edge_msg_kernel<<<edgeBlocks, TB, 0, stream>>>(Vf, erow, ecol, scores, mmax, denom, agg, E);
        agg_norm_kernel<<<ceil_div((long long)NH, TB), TB, 0, stream>>>(agg, denom, aggb, N);

        // MLP on cat([h, agg]): split-K as two GEMMs into t1 (=Qf)
        gemm4(hb,   mw1b + w1Off,                        mlp_b1 + bOff, nullptr, Qf, nullptr, N, HIDC, HIDC, 0);
        gemm4(aggb, mw1b + w1Off + (size_t)HIDC * HIDC,  nullptr,       nullptr, Qf, nullptr, N, HIDC, HIDC, EP_ACCUM | EP_RELU);
        ln_bf16_kernel<<<lnBlocksN, TB, 0, stream>>>(Qf, mlp_g + bOff, mlp_beta + bOff, t1b, N);
        // z @ W2 + b2 + h  ->  h (and hb)
        gemm4(t1b, mw2b + wOff, mlp_b2 + bOff, h, h, hb, N, HIDC, HIDC, 0);
    }

    // ---- global mean pool + classifier ----
    zero(pooled, (size_t)NGRAPHS * HIDC);
    zero(cnt, NGRAPHS);
    pool_sum_kernel<<<ceil_div((long long)NH, TB), TB, 0, stream>>>(h, batch, pooled, cnt, N);
    pool_finish_kernel<<<ceil_div(NGRAPHS * HIDC, TB), TB, 0, stream>>>(pooled, cnt, pooledb);

    gemm4(pooledb, cw1b, cls_b1, nullptr, c1, nullptr, NGRAPHS, HIDC, HIDC, EP_RELU);
    ln_bf16_kernel<<<ceil_div(NGRAPHS, 8), TB, 0, stream>>>(c1, cls_g, cls_beta, c1b, NGRAPHS);
    {
        const int blocks = ceil_div(NGRAPHS, 256) * (OUTC / 32);
        wmma_gemm_lds_kernel<2><<<blocks, TB, 0, stream>>>(
            c1b, cw2b, cls_b2, nullptr, (float*)d_out, nullptr, NGRAPHS, HIDC, OUTC, 0);
    }
}
